// AdaptiveEMAModel_979252544296
// MI455X (gfx1250) — matrix operand order repaired and checked by reading, not compiled, not run on hardware
//
#include <hip/hip_runtime.h>
#include <hip/hip_bf16.h>

typedef __attribute__((ext_vector_type(16))) __bf16 v16bf;
typedef __attribute__((ext_vector_type(8)))  float  v8f;

#define HDIM 128
#define FDIM 256
#define SEQL 2048
#define BATCH 64
#define VOCAB 32000

#if __has_builtin(__builtin_amdgcn_global_load_async_to_lds_b128) && \
    __has_builtin(__builtin_amdgcn_s_wait_asynccnt)
#define SCAN_ASYNC 1
typedef int v4i_gcc __attribute__((vector_size(16)));
typedef __attribute__((address_space(1))) v4i_gcc* as1_v4i;   // global side
typedef __attribute__((address_space(3))) v4i_gcc* as3_v4i;   // LDS side
#else
#define SCAN_ASYNC 0
#endif

// ---------------- WMMA fragment loaders (contiguous k = 16*(lane/16) + e in A and B, wave32) --------

// A-matrix 16x32 bf16, row-major source with row stride ld.
__device__ __forceinline__ v16bf load_a_frag(const __bf16* A, int ld, int k0, int lane) {
    int row   = lane & 15;
    int kbase = (lane < 16) ? 0 : 16;
    const __bf16* p = A + row * ld + k0 + kbase;
    v16bf a;
#pragma unroll
    for (int e = 0; e < 16; ++e) a[e] = p[e];
    return a;
}

// B-matrix 32x16 bf16 where B[k][n] = W[n0+n][k0+k], W row-major stride ldw.
__device__ __forceinline__ v16bf load_b_fragT(const __bf16* W, int ldw, int n0, int k0, int lane) {
    int col   = lane & 15;
    int kbase = (lane < 16) ? 0 : 16;
    const __bf16* p = W + (size_t)(n0 + col) * ldw + k0 + kbase;
    v16bf b;
#pragma unroll
    for (int e = 0; e < 16; ++e) b[e] = p[e];
    return b;
}

// ---------------- weight fp32 -> bf16 ----------------------------------------
__global__ void cvt_kernel(const float* __restrict__ s, __bf16* __restrict__ d, int n) {
    int i = blockIdx.x * blockDim.x + threadIdx.x;
    if (i < n) d[i] = (__bf16)s[i];
}

// ---------------- fused embed + FFN + LN + k-projection ----------------------
// one block = 16 rows of (B*L), 128 threads = 4 waves
__global__ __launch_bounds__(128) void fused_ffn_kernel(
    const int* __restrict__ seq, const float* __restrict__ embed_w,
    const __bf16* __restrict__ w1b, const float* __restrict__ b1,
    const __bf16* __restrict__ w2b, const float* __restrict__ b2,
    const float* __restrict__ ln_g, const float* __restrict__ ln_b,
    const __bf16* __restrict__ kpb, float* __restrict__ kall) {
    __shared__ int    tok[16];
    __shared__ float  hf[16][HDIM];
    __shared__ __bf16 hb[16][HDIM];
    __shared__ __bf16 f1b[16][FDIM];
    __shared__ float  xf[16][HDIM];
    __shared__ __bf16 hlnb[16][HDIM];
    __shared__ float  ps[16][8], ps2[16][8];
    __shared__ float  mu[16], rs[16];

    const int tid  = threadIdx.x;
    const int r0   = blockIdx.x * 16;
    const int wave = tid >> 5;
    const int lane = tid & 31;

    if (tid < 16) tok[tid] = seq[r0 + tid];
    __syncthreads();

#pragma unroll 4
    for (int i = 0; i < 16; ++i) {
        float v = embed_w[(size_t)tok[i] * HDIM + tid];
        hf[i][tid] = v;
        hb[i][tid] = (__bf16)v;
    }
    __syncthreads();

    // GEMM1: (16x128) @ W1^T -> 16x256, ReLU
    for (int j = 0; j < 4; ++j) {
        int n0 = (wave * 4 + j) * 16;
        v8f acc = {};
#pragma unroll
        for (int ks = 0; ks < 4; ++ks) {
            v16bf a  = load_a_frag(&hb[0][0], HDIM, ks * 32, lane);
            v16bf bb = load_b_fragT(w1b, HDIM, n0, ks * 32, lane);
            acc = __builtin_amdgcn_wmma_f32_16x16x32_bf16(false, a, false, bb,
                                                          (short)0, acc, false, false);
        }
        int col = n0 + (lane & 15);
        float bias = b1[col];
#pragma unroll
        for (int rr = 0; rr < 8; ++rr) {
            int row = (lane < 16) ? rr : rr + 8;
            float v = acc[rr] + bias;
            f1b[row][col] = (__bf16)(v > 0.f ? v : 0.f);
        }
    }
    __syncthreads();

    // GEMM2: (16x256) @ W2^T -> 16x128, + bias + residual
    for (int j = 0; j < 2; ++j) {
        int n0 = (wave * 2 + j) * 16;
        v8f acc = {};
#pragma unroll
        for (int ks = 0; ks < 8; ++ks) {
            v16bf a  = load_a_frag(&f1b[0][0], FDIM, ks * 32, lane);
            v16bf bb = load_b_fragT(w2b, FDIM, n0, ks * 32, lane);
            acc = __builtin_amdgcn_wmma_f32_16x16x32_bf16(false, a, false, bb,
                                                          (short)0, acc, false, false);
        }
        int col = n0 + (lane & 15);
        float bias = b2[col];
#pragma unroll
        for (int rr = 0; rr < 8; ++rr) {
            int row = (lane < 16) ? rr : rr + 8;
            xf[row][col] = acc[rr] + bias + hf[row][col];
        }
    }
    __syncthreads();

    // LayerNorm (population variance, eps=1e-5)
    {
        int i = tid >> 3, s = tid & 7;
        float sum = 0.f, sq = 0.f;
#pragma unroll
        for (int cc = 0; cc < 16; ++cc) {
            float v = xf[i][s * 16 + cc];
            sum += v; sq += v * v;
        }
        ps[i][s] = sum; ps2[i][s] = sq;
    }
    __syncthreads();
    if (tid < 16) {
        float sum = 0.f, sq = 0.f;
#pragma unroll
        for (int s = 0; s < 8; ++s) { sum += ps[tid][s]; sq += ps2[tid][s]; }
        float m = sum * (1.0f / HDIM);
        float var = sq * (1.0f / HDIM) - m * m;
        mu[tid] = m;
        rs[tid] = rsqrtf(var + 1e-5f);
    }
    __syncthreads();
    {
        int i = tid >> 3, s = tid & 7;
        float m = mu[i], rsd = rs[i];
#pragma unroll
        for (int cc = 0; cc < 16; ++cc) {
            int c = s * 16 + cc;
            hlnb[i][c] = (__bf16)((xf[i][c] - m) * rsd * ln_g[c] + ln_b[c]);
        }
    }
    __syncthreads();

    // k-projection: (16x128) @ kp^T -> 16x128 (f32 out)
    for (int j = 0; j < 2; ++j) {
        int n0 = (wave * 2 + j) * 16;
        v8f acc = {};
#pragma unroll
        for (int ks = 0; ks < 4; ++ks) {
            v16bf a  = load_a_frag(&hlnb[0][0], HDIM, ks * 32, lane);
            v16bf bb = load_b_fragT(kpb, HDIM, n0, ks * 32, lane);
            acc = __builtin_amdgcn_wmma_f32_16x16x32_bf16(false, a, false, bb,
                                                          (short)0, acc, false, false);
        }
        int col = n0 + (lane & 15);
#pragma unroll
        for (int rr = 0; rr < 8; ++rr) {
            int row = (lane < 16) ? rr : rr + 8;
            kall[(size_t)(r0 + row) * HDIM + col] = acc[rr];
        }
    }
}

// ---------------- sequential delta-rule scan, M resident in LDS --------------
// one block per batch, 512 threads = 16 waves, 4 threads per M-row.
// Wave32 shuffle-butterfly reductions; k rows double-buffered in LDS via
// GLOBAL_LOAD_ASYNC_TO_LDS_B128 (ASYNCcnt) when available.
#define MSTRIDE 132   // 528B row pitch; 32-float segments 16B aligned
__global__ __launch_bounds__(512) void scan_kernel(
    const float* __restrict__ kall, const float* __restrict__ rp_w,
    const float* __restrict__ rp_b, __bf16* __restrict__ rd2) {
    extern __shared__ float sm[];
    float* M   = sm;                   // 128*132
    float* kvb = M + 128 * MSTRIDE;    // 2 x 128 (double-buffered k row)
    float* kn  = kvb + 256;            // 128
    float* red = kn + 128;             // 32 (16 partials x 2 buffers)

    const int tid  = threadIdx.x;
    const int b    = blockIdx.x;
    const int wv   = tid >> 5;
    const int lane = tid & 31;
    const int r    = tid >> 2;          // M row (0..127)
    const int c0   = (tid & 3) * 32;    // column quarter
    const float coeff = 1.0f / (float)SEQL;   // (1 - alpha) = 1/L

    for (int idx = tid; idx < 128 * MSTRIDE; idx += 512) M[idx] = 0.f;

    const float* kbat = kall + (size_t)b * SEQL * HDIM;

#if SCAN_ASYNC
    // preload k row 0 into buffer 0 (one b128 per lane of wave 0 = 512B row)
    if (wv == 0)
        __builtin_amdgcn_global_load_async_to_lds_b128(
            (as1_v4i)(kbat + lane * 4), (as3_v4i)(kvb + lane * 4), 0, 0);
#endif
    __syncthreads();

    for (int t = 0; t < SEQL - 1; ++t) {
        float* kv = kvb + ((t & 1) << 7);
#if SCAN_ASYNC
        if (wv == 0) __builtin_amdgcn_s_wait_asynccnt(0);
        __syncthreads();               // B0: row t visible; step t-1 fully done
        if (wv == 0)                   // overlap row t+1 copy with this step
            __builtin_amdgcn_global_load_async_to_lds_b128(
                (as1_v4i)(kbat + (size_t)(t + 1) * HDIM + lane * 4),
                (as3_v4i)(kvb + (((t + 1) & 1) << 7) + lane * 4), 0, 0);
        float kreg = 0.f, val = 0.f;
        if (tid < 128) {
            kreg = kv[tid];
            val  = kreg * kreg;
        }
#else
        const float* kptr = kbat + (size_t)t * HDIM;
        if (tid < 4) __builtin_prefetch(kptr + HDIM + tid * 32, 0, 0);
        float kreg = 0.f, val = 0.f;
        if (tid < 128) {
            kreg = kptr[tid];
            kv[tid] = kreg;
            val = kreg * kreg;
        }
#endif
#pragma unroll
        for (int m = 16; m >= 1; m >>= 1) val += __shfl_xor(val, m, 32);
        if (lane == 0) red[wv] = val;
        __syncthreads();                          // B1: kv + red[0..15]

        float kn2 = 0.f;
#pragma unroll
        for (int w = 0; w < 16; ++w) kn2 += red[w];
        float rinv = 1.0f / fmaxf(sqrtf(kn2), 1e-12f);
        if (tid < 128) kn[tid] = kreg * rinv;
        __syncthreads();                          // B2: kn

        // vp = M @ kn  (4 lanes per row, quad combine via shuffles)
        float s_ = 0.f;
        {
            const float* Mr = M + r * MSTRIDE + c0;
            const float* kp = kn + c0;
#pragma unroll 8
            for (int c = 0; c < 32; ++c) s_ = fmaf(Mr[c], kp[c], s_);
        }
        s_ += __shfl_xor(s_, 1, 32);
        s_ += __shfl_xor(s_, 2, 32);
        float d_r = kv[r] - s_;                   // all 4 quad lanes hold d[r]

        float v2 = ((tid & 3) == 0) ? d_r * d_r : 0.f;
#pragma unroll
        for (int m = 16; m >= 1; m >>= 1) v2 += __shfl_xor(v2, m, 32);
        if (lane == 0) red[16 + wv] = v2;
        __syncthreads();                          // B3: red[16..31]

        float dn2 = 0.f;
#pragma unroll
        for (int w = 0; w < 16; ++w) dn2 += red[16 + w];

        if (dn2 >= 0.16f * kn2) {                 // fire: ||d|| >= 0.4*||k||
            float dr = d_r * coeff;
            float* Mw = M + r * MSTRIDE + c0;
            const float* kp = kn + c0;
#pragma unroll 8
            for (int c = 0; c < 32; ++c) Mw[c] = fmaf(dr, kp[c], Mw[c]);
        }
#if !SCAN_ASYNC
        __syncthreads();               // B4 (async path: next B0 subsumes it)
#endif
    }

    // read = M @ q ; rd2 = read @ rp^T + rp_b  (bf16 for the final WMMA GEMM)
#if SCAN_ASYNC
    if (wv == 0) __builtin_amdgcn_s_wait_asynccnt(0);
    __syncthreads();
    const float* qv = kvb + (((SEQL - 1) & 1) << 7);   // row L-1 already in LDS
#else
    const float* qptr = kbat + (size_t)(SEQL - 1) * HDIM;
    if (tid < 128) kvb[tid] = qptr[tid];
    __syncthreads();
    const float* qv = kvb;
#endif
    {
        float s_ = 0.f;
        const float* Mr = M + r * MSTRIDE + c0;
        const float* kp = qv + c0;
#pragma unroll 8
        for (int c = 0; c < 32; ++c) s_ = fmaf(Mr[c], kp[c], s_);
        s_ += __shfl_xor(s_, 1, 32);
        s_ += __shfl_xor(s_, 2, 32);
        if ((tid & 3) == 0) kn[r] = s_;           // kn reused as read-vector
    }
    __syncthreads();
    {
        float s_ = 0.f;
        const float* W  = rp_w + (size_t)r * HDIM + c0;
        const float* rv = kn + c0;
#pragma unroll 8
        for (int c = 0; c < 32; ++c) s_ = fmaf(rv[c], W[c], s_);
        s_ += __shfl_xor(s_, 1, 32);
        s_ += __shfl_xor(s_, 2, 32);
        if ((tid & 3) == 0)
            rd2[b * HDIM + r] = (__bf16)(s_ + rp_b[r]);
    }
}

// ---------------- logits GEMM: (64x128) @ out_w^T -> (64x32000) --------------
__global__ __launch_bounds__(256) void logits_kernel(
    const __bf16* __restrict__ rd2, const __bf16* __restrict__ outw,
    const float* __restrict__ out_b, float* __restrict__ out) {
    __shared__ __bf16 ab[16 * HDIM];
    const int tid = threadIdx.x;
    const int m0  = blockIdx.y * 16;
    const int g0  = blockIdx.x * 128;

    for (int idx = tid; idx < 16 * HDIM; idx += 256)
        ab[idx] = rd2[(m0 + (idx >> 7)) * HDIM + (idx & 127)];
    __syncthreads();

    const int wave = tid >> 5, lane = tid & 31;
    const int col0 = g0 + wave * 16;
    v8f acc = {};
#pragma unroll
    for (int ks = 0; ks < 4; ++ks) {
        v16bf a  = load_a_frag(ab, HDIM, ks * 32, lane);
        v16bf bb = load_b_fragT(outw, HDIM, col0, ks * 32, lane);
        if (ks < 3)
            __builtin_prefetch(outw + (size_t)(col0 + (lane & 15)) * HDIM + (ks + 1) * 32, 0, 1);
        acc = __builtin_amdgcn_wmma_f32_16x16x32_bf16(false, a, false, bb,
                                                      (short)0, acc, false, false);
    }
    const int col = col0 + (lane & 15);
    const float bias = out_b[col];
#pragma unroll
    for (int rr = 0; rr < 8; ++rr) {
        int row = m0 + ((lane < 16) ? rr : rr + 8);
        out[(size_t)row * VOCAB + col] = acc[rr] + bias;
    }
}

// ---------------- launch ------------------------------------------------------
extern "C" void kernel_launch(void* const* d_in, const int* in_sizes, int n_in,
                              void* d_out, int out_size, void* d_ws, size_t ws_size,
                              hipStream_t stream) {
    const int*   seq     = (const int*)d_in[0];
    const float* embed_w = (const float*)d_in[1];
    const float* ff_w1   = (const float*)d_in[2];
    const float* ff_b1   = (const float*)d_in[3];
    const float* ff_w2   = (const float*)d_in[4];
    const float* ff_b2   = (const float*)d_in[5];
    const float* ln_g    = (const float*)d_in[6];
    const float* ln_b    = (const float*)d_in[7];
    const float* kp_w    = (const float*)d_in[8];
    const float* rp_w    = (const float*)d_in[9];
    const float* rp_b    = (const float*)d_in[10];
    const float* out_w   = (const float*)d_in[11];
    const float* out_b   = (const float*)d_in[12];
    float*       out     = (float*)d_out;

    char* ws = (char*)d_ws;
    float*  kall  = (float*)(ws + 0);                               // 64 MB
    __bf16* w1b   = (__bf16*)(ws + 67108864);                       // 256*128
    __bf16* w2b   = (__bf16*)(ws + 67108864 + 65536);               // 128*256
    __bf16* kpb   = (__bf16*)(ws + 67108864 + 131072);              // 128*128
    __bf16* outwb = (__bf16*)(ws + 67108864 + 163840);              // 32000*128
    __bf16* rd2   = (__bf16*)(ws + 67108864 + 163840 + 8192000);    // 64*128

    cvt_kernel<<<(FDIM * HDIM + 255) / 256, 256, 0, stream>>>(ff_w1, w1b, FDIM * HDIM);
    cvt_kernel<<<(HDIM * FDIM + 255) / 256, 256, 0, stream>>>(ff_w2, w2b, HDIM * FDIM);
    cvt_kernel<<<(HDIM * HDIM + 255) / 256, 256, 0, stream>>>(kp_w, kpb, HDIM * HDIM);
    cvt_kernel<<<(VOCAB * HDIM + 255) / 256, 256, 0, stream>>>(out_w, outwb, VOCAB * HDIM);

    fused_ffn_kernel<<<(BATCH * SEQL) / 16, 128, 0, stream>>>(
        seq, embed_w, w1b, ff_b1, w2b, ff_b2, ln_g, ln_b, kpb, kall);

    size_t shmem = (size_t)(128 * MSTRIDE + 256 + 128 + 32) * sizeof(float);
    scan_kernel<<<BATCH, 512, shmem, stream>>>(kall, rp_w, rp_b, rd2);

    logits_kernel<<<dim3(VOCAB / 128, BATCH / 16), 256, 0, stream>>>(rd2, outwb, out_b, out);
}
